// GCNLP_32315333935772
// MI455X (gfx1250) — matrix operand order repaired and checked by reading, not compiled, not run on hardware
//
#include <hip/hip_runtime.h>
#include <hip/hip_bf16.h>

typedef __attribute__((ext_vector_type(16))) __bf16 v16bf;
typedef __attribute__((ext_vector_type(8)))  float  v8f;

// ---------------- init: deg=1 (self loop), agg buffers = 0 ----------------
__global__ void k_init(float* __restrict__ inv, float* __restrict__ hagg,
                       float* __restrict__ zagg, int N) {
  long long t = (long long)blockIdx.x * blockDim.x + threadIdx.x;
  long long n32 = 32LL * N, n16 = 16LL * N;
  if (t < N)                       inv[t] = 1.0f;
  else if (t < (long long)N + n32) hagg[t - N] = 0.0f;
  else if (t < (long long)N + n32 + n16) zagg[t - N - n32] = 0.0f;
}

// ---------------- degree accumulation ----------------
__global__ void k_degree(const int* __restrict__ dst, float* __restrict__ deg, int E) {
  int e = blockIdx.x * blockDim.x + threadIdx.x;
  if (e < E) unsafeAtomicAdd(&deg[dst[e]], 1.0f);
}

__global__ void k_rsqrt(float* __restrict__ v, int N) {
  int i = blockIdx.x * blockDim.x + threadIdx.x;
  if (i < N) v[i] = rsqrtf(v[i]);
}

// ---------------- GEMM1: xw[N,32] = x[N,128] @ W1[128,32]  (bf16 WMMA) ----
__global__ void k_gemm1(const float* __restrict__ x, const float* __restrict__ W1,
                        float* __restrict__ xw, int n_tiles) {
  const int lane = threadIdx.x & 31;
  const int wid  = threadIdx.x >> 5;
  const int lm   = lane & 15;       // row (A) / col (B/D) within tile
  const int half = lane >> 4;       // K-half selector

  // Preload B fragments: 4 k-steps x 2 n-tiles. B: lane holds col n, K=half*16+j.
  v16bf Bf[4][2];
#pragma unroll
  for (int s = 0; s < 4; ++s)
#pragma unroll
    for (int nt = 0; nt < 2; ++nt) {
      int n = nt * 16 + lm;
#pragma unroll
      for (int j = 0; j < 16; ++j)
        Bf[s][nt][j] = (__bf16)W1[(s * 32 + half * 16 + j) * 32 + n];
    }

  const int wavesTotal = gridDim.x * (blockDim.x >> 5);
  for (int tile = blockIdx.x * (blockDim.x >> 5) + wid; tile < n_tiles; tile += wavesTotal) {
    const float* xr = x + (long long)(tile * 16 + lm) * 128;
    v8f c0 = {}, c1 = {};
#pragma unroll
    for (int s = 0; s < 4; ++s) {
      // A: lane holds row m; K=half*16+j
      v16bf A;
#pragma unroll
      for (int j = 0; j < 8; ++j) A[j]     = (__bf16)xr[s * 32 + half * 16 + j];
#pragma unroll
      for (int j = 0; j < 8; ++j) A[8 + j] = (__bf16)xr[s * 32 + half * 16 + 8 + j];
      c0 = __builtin_amdgcn_wmma_f32_16x16x32_bf16(false, A, false, Bf[s][0], (short)0, c0, false, false);
      c1 = __builtin_amdgcn_wmma_f32_16x16x32_bf16(false, A, false, Bf[s][1], (short)0, c1, false, false);
    }
    // D: element r -> row r + 8*half, col lm (c0: n=lm, c1: n=16+lm)
    int mbase = tile * 16 + 8 * half;
#pragma unroll
    for (int r = 0; r < 8; ++r) {
      xw[(long long)(mbase + r) * 32 + lm]      = c0[r];
      xw[(long long)(mbase + r) * 32 + 16 + lm] = c1[r];
    }
  }
}

// ---------------- GEMM2: hw[N,16] = h[N,32] @ W2[32,16]  (bf16 WMMA) -----
__global__ void k_gemm2(const float* __restrict__ h, const float* __restrict__ W2,
                        float* __restrict__ hw, int n_tiles) {
  const int lane = threadIdx.x & 31;
  const int wid  = threadIdx.x >> 5;
  const int lm   = lane & 15;
  const int half = lane >> 4;

  v16bf B;
#pragma unroll
  for (int j = 0; j < 16; ++j) B[j] = (__bf16)W2[(half * 16 + j) * 16 + lm];

  const int wavesTotal = gridDim.x * (blockDim.x >> 5);
  for (int tile = blockIdx.x * (blockDim.x >> 5) + wid; tile < n_tiles; tile += wavesTotal) {
    const float* hr = h + (long long)(tile * 16 + lm) * 32;
    v16bf A;
#pragma unroll
    for (int j = 0; j < 8; ++j) A[j]     = (__bf16)hr[half * 16 + j];
#pragma unroll
    for (int j = 0; j < 8; ++j) A[8 + j] = (__bf16)hr[half * 16 + 8 + j];
    v8f c = {};
    c = __builtin_amdgcn_wmma_f32_16x16x32_bf16(false, A, false, B, (short)0, c, false, false);
    int mbase = tile * 16 + 8 * half;
#pragma unroll
    for (int r = 0; r < 8; ++r)
      hw[(long long)(mbase + r) * 16 + lm] = c[r];
  }
}

// ---------------- edge aggregation: fout[d,:] += norm * fin[s,:] ----------
template <int FSHIFT>
__global__ void k_agg(const int* __restrict__ src, const int* __restrict__ dst,
                      const float* __restrict__ inv, const float* __restrict__ fin,
                      float* __restrict__ fout, int E) {
  const int F = 1 << FSHIFT;
  long long t = (long long)blockIdx.x * blockDim.x + threadIdx.x;
  if (t >= ((long long)E << FSHIFT)) return;
  int e = (int)(t >> FSHIFT);
  int f = (int)(t & (F - 1));
  int s = src[e], d = dst[e];
  float w = inv[s] * inv[d];
  unsafeAtomicAdd(&fout[(long long)d * F + f], fin[(long long)s * F + f] * w);
}

// ---------------- epilogue 1: h = relu(hagg + inv^2*xw + b1) (in place) ---
__global__ void k_bias_relu(float* __restrict__ h, const float* __restrict__ xw,
                            const float* __restrict__ inv, const float* __restrict__ b, int N) {
  int t = blockIdx.x * blockDim.x + threadIdx.x;
  if (t >= N * 32) return;
  int i = t >> 5, f = t & 31;
  float s = inv[i];
  float v = h[t] + s * s * xw[t] + b[f];
  h[t] = v > 0.0f ? v : 0.0f;
}

// ---------------- epilogue 2: z = zagg + inv^2*hw + b2 (in place) ---------
__global__ void k_bias2(float* __restrict__ z, const float* __restrict__ hw,
                        const float* __restrict__ inv, const float* __restrict__ b, int N) {
  int t = blockIdx.x * blockDim.x + threadIdx.x;
  if (t >= N * 16) return;
  int i = t >> 4, f = t & 15;
  float s = inv[i];
  z[t] = z[t] + s * s * hw[t] + b[f];
}

// ---------------- decode: logits[e] = dot(z[a], z[b]) over 16 dims --------
__global__ void k_decode(const int* __restrict__ eli, const float* __restrict__ z,
                         float* __restrict__ out, int L) {
  int e = blockIdx.x * blockDim.x + threadIdx.x;
  if (e >= L) return;
  int a = eli[e], b = eli[L + e];
  const float4* za = reinterpret_cast<const float4*>(z) + (long long)a * 4;
  const float4* zb = reinterpret_cast<const float4*>(z) + (long long)b * 4;
  float acc = 0.0f;
#pragma unroll
  for (int q = 0; q < 4; ++q) {
    float4 u = za[q], v = zb[q];
    acc += u.x * v.x + u.y * v.y + u.z * v.z + u.w * v.w;
  }
  out[e] = acc;
}

extern "C" void kernel_launch(void* const* d_in, const int* in_sizes, int n_in,
                              void* d_out, int out_size, void* d_ws, size_t ws_size,
                              hipStream_t stream) {
  const float* x  = (const float*)d_in[0];
  const int*   ei = (const int*)d_in[1];
  const int*   el = (const int*)d_in[2];
  const float* W1 = (const float*)d_in[3];
  const float* b1 = (const float*)d_in[4];
  const float* W2 = (const float*)d_in[5];
  const float* b2 = (const float*)d_in[6];
  float* out = (float*)d_out;

  const int N = in_sizes[0] / 128;     // 100000
  const int E = in_sizes[1] / 2;       // 3200000
  const int L = in_sizes[2] / 2;       // 200000
  const int* src = ei;
  const int* dst = ei + E;

  float* ws  = (float*)d_ws;
  float* inv = ws;                          // N      (deg -> inv_sqrt)
  float* xw  = ws + (long long)N;           // 32N
  float* h   = ws + 33LL * N;               // 32N    (agg -> relu'd, in place)
  float* hw  = ws + 65LL * N;               // 16N
  float* z   = ws + 81LL * N;               // 16N    (agg -> final, in place)

  const int tpb = 256;
  long long initTot = 49LL * N;
  k_init<<<(int)((initTot + tpb - 1) / tpb), tpb, 0, stream>>>(inv, h, z, N);
  k_degree<<<(E + tpb - 1) / tpb, tpb, 0, stream>>>(dst, inv, E);
  k_rsqrt<<<(N + tpb - 1) / tpb, tpb, 0, stream>>>(inv, N);

  const int n_tiles = N / 16;               // 6250 (N divisible by 16)
  k_gemm1<<<208, 256, 0, stream>>>(x, W1, xw, n_tiles);

  long long a1 = (long long)E * 32;
  k_agg<5><<<(int)((a1 + tpb - 1) / tpb), tpb, 0, stream>>>(src, dst, inv, xw, h, E);
  k_bias_relu<<<(N * 32 + tpb - 1) / tpb, tpb, 0, stream>>>(h, xw, inv, b1, N);

  k_gemm2<<<208, 256, 0, stream>>>(h, W2, hw, n_tiles);

  long long a2 = (long long)E * 16;
  k_agg<4><<<(int)((a2 + tpb - 1) / tpb), tpb, 0, stream>>>(src, dst, inv, hw, z, E);
  k_bias2<<<(N * 16 + tpb - 1) / tpb, tpb, 0, stream>>>(z, hw, inv, b2, N);

  k_decode<<<(L + tpb - 1) / tpb, tpb, 0, stream>>>(el, z, out, L);
}
